// BasicAttn_90967407329826
// MI455X (gfx1250) — matrix operand order repaired and checked by reading, not compiled, not run on hardware
//
#include <hip/hip_runtime.h>

typedef __attribute__((ext_vector_type(16))) _Float16 v16h;
typedef __attribute__((ext_vector_type(8)))  _Float16 v8h;
typedef __attribute__((ext_vector_type(4)))  _Float16 v4h;
typedef __attribute__((ext_vector_type(8)))  float    v8f;
typedef __attribute__((ext_vector_type(4)))  float    v4f;

#define SEQ 2048
#define DKH 64
#define QT  32          // query rows per workgroup (2 row-blocks of 16)
#define NWAVE 8

__global__ __launch_bounds__(256) void
attn_fwd_kernel(const float* __restrict__ Qg,
                const float* __restrict__ Kg,
                const float* __restrict__ Vg,
                float* __restrict__ ctx,
                float* __restrict__ attn)
{
    // 32 x 2048 f16 tile: scores (pass 1) then probabilities (pass 2). 128 KB.
    __shared__ __align__(16) _Float16 s_p[QT * SEQ];

    const int tid  = threadIdx.x;
    const int lane = tid & 31;
    const int wave = tid >> 5;
    const int sel  = lane >> 4;   // wave half (0/1)
    const int l16  = lane & 15;

    const int bh = blockIdx.y;            // b*H + h
    const int q0 = blockIdx.x * QT;       // first query row of this tile

    const size_t base = (size_t)bh * SEQ * DKH;
    const float* Qb = Qg + base;
    const float* Kb = Kg + base;
    const float* Vb = Vg + base;

    // ---------------- Pass 1: S = (Q K^T) * 1/sqrt(dk), f16 -> LDS ----------
    // A 16x32 f16 layout: lane -> M = lane&15; halves e<8 : K = sel*8+e,
    //                                          e>=8: K = sel*8+16+(e-8).
    // Two query row-blocks -> 4 A fragments, loaded once per wave.
    v16h aq[2][2];
    #pragma unroll
    for (int rb = 0; rb < 2; ++rb) {
        const float* qrow = Qb + (size_t)(q0 + rb * 16 + l16) * DKH;
        #pragma unroll
        for (int kc = 0; kc < 2; ++kc) {
            const int k = kc * 32 + sel * 8;
            v4f f0 = *(const v4f*)(qrow + k);
            v4f f1 = *(const v4f*)(qrow + k + 4);
            v4f f2 = *(const v4f*)(qrow + k + 16);
            v4f f3 = *(const v4f*)(qrow + k + 20);
            v16h a;
            #pragma unroll
            for (int j = 0; j < 4; ++j) {
                a[j]      = (_Float16)f0[j];
                a[4 + j]  = (_Float16)f1[j];
                a[8 + j]  = (_Float16)f2[j];
                a[12 + j] = (_Float16)f3[j];
            }
            aq[rb][kc] = a;
        }
    }

    // 128 key tiles of 16 columns, 16 tiles per wave; each K B-fragment
    // feeds two WMMAs (both query row-blocks).
    for (int t = wave; t < SEQ / 16; t += NWAVE) {
        const int n0 = t * 16;
        const float* krow = Kb + (size_t)(n0 + l16) * DKH;
        if (t + NWAVE < SEQ / 16)
            __builtin_prefetch(krow + (size_t)NWAVE * 16 * DKH, 0, 1);

        // B 32x16 f16 layout: lane -> N = lane&15; e<8: K = sel*8+e, e>=8: K = sel*8+16+(e-8).
        v16h bk[2];
        #pragma unroll
        for (int kc = 0; kc < 2; ++kc) {
            const int k = kc * 32 + sel * 8;
            v4f f0 = *(const v4f*)(krow + k);
            v4f f1 = *(const v4f*)(krow + k + 4);
            v4f f2 = *(const v4f*)(krow + k + 16);
            v4f f3 = *(const v4f*)(krow + k + 20);
            v16h b;
            #pragma unroll
            for (int j = 0; j < 4; ++j) {
                b[j]      = (_Float16)f0[j];
                b[4 + j]  = (_Float16)f1[j];
                b[8 + j]  = (_Float16)f2[j];
                b[12 + j] = (_Float16)f3[j];
            }
            bk[kc] = b;
        }

        const float scale = 0.125f;  // 1/sqrt(64)
        #pragma unroll
        for (int rb = 0; rb < 2; ++rb) {
            v8f d = {};
            d = __builtin_amdgcn_wmma_f32_16x16x32_f16(false, aq[rb][0],
                                                       false, bk[0],
                                                       (short)0, d, false, false);
            d = __builtin_amdgcn_wmma_f32_16x16x32_f16(false, aq[rb][1],
                                                       false, bk[1],
                                                       (short)0, d, false, false);
            // C/D layout: VGPR r -> M = r + sel*8, N = lane&15.
            #pragma unroll
            for (int r = 0; r < 8; ++r)
                s_p[(rb * 16 + sel * 8 + r) * SEQ + n0 + l16] =
                    (_Float16)(d[r] * scale);
        }
    }
    __syncthreads();

    // ---------------- Softmax (fp32) + write attn to HBM --------------------
    // Wave w owns rows 4w..4w+3. Row cached in registers as 16 x v4h.
    // exp() evaluated once per element (cached back into the fragments).
    // attn stores: non-temporal v4f at col = lane*4 + j*128 -> dense 512B
    // wavefront stores that bypass L2 retention (write-once data).
    {
        #pragma unroll
        for (int rr = 0; rr < 4; ++rr) {
            const int row = wave * 4 + rr;
            _Float16* srow = &s_p[row * SEQ];
            v4h frag[16];
            float mx = -3.0e38f;
            #pragma unroll
            for (int j = 0; j < 16; ++j) {
                frag[j] = *(const v4h*)&srow[lane * 4 + j * 128];
                #pragma unroll
                for (int e = 0; e < 4; ++e)
                    mx = fmaxf(mx, (float)frag[j][e]);
            }
            #pragma unroll
            for (int off = 16; off > 0; off >>= 1)
                mx = fmaxf(mx, __shfl_xor(mx, off, 32));

            float sum = 0.0f;
            #pragma unroll
            for (int j = 0; j < 16; ++j) {
                #pragma unroll
                for (int e = 0; e < 4; ++e) {
                    const float ex = __expf((float)frag[j][e] - mx);
                    sum += ex;
                    frag[j][e] = (_Float16)ex;   // cache exp, reuse below
                }
            }
            #pragma unroll
            for (int off = 16; off > 0; off >>= 1)
                sum += __shfl_xor(sum, off, 32);
            const float inv = 1.0f / sum;

            float* arow = attn + ((size_t)bh * SEQ + (q0 + row)) * SEQ;
            #pragma unroll
            for (int j = 0; j < 16; ++j) {
                v4f p;
                v4h ph;
                #pragma unroll
                for (int e = 0; e < 4; ++e) {
                    p[e]  = (float)frag[j][e] * inv;
                    ph[e] = (_Float16)p[e];
                }
                __builtin_nontemporal_store(p, (v4f*)&arow[lane * 4 + j * 128]);
                *(v4h*)&srow[lane * 4 + j * 128] = ph;  // f16 probs for pass 2
            }
        }
    }
    __syncthreads();

    // ---------------- Pass 2: context = P @ V ------------------------------
    // 8 output tiles (2 row-blocks x 4 col-tiles) == 8 waves; each wave owns
    // one 16x16 context tile over the full K = 2048 (64 WMMAs), no reduction.
    {
        const int rowblk  = wave >> 2;
        const int coltile = wave & 3;
        const int m0 = rowblk * 16;
        const int n0 = coltile * 16;
        v8f acc = {};
        for (int kc = 0; kc < SEQ / 32; ++kc) {
            // A fragment (P) from LDS: two 16B ds loads per lane.
            const int kb = kc * 32 + sel * 8;
            v8h lo = *(const v8h*)&s_p[(m0 + l16) * SEQ + kb];
            v8h hi = *(const v8h*)&s_p[(m0 + l16) * SEQ + kb + 16];
            v16h a;
            #pragma unroll
            for (int j = 0; j < 8; ++j) { a[j] = lo[j]; a[8 + j] = hi[j]; }

            // B fragment from V (fp32 -> f16); K row = kc*32 + sel*8 + e (e<8),
            // kc*32 + sel*8 + 16 + (e-8) (e>=8), column n0 + (lane&15).
            const int kk = kc * 32 + sel * 8;
            const float* vcol = Vb + (size_t)kk * DKH + n0 + l16;
            v16h b;
            #pragma unroll
            for (int e = 0; e < 16; ++e)
                b[e] = (_Float16)vcol[(size_t)(e + (e & 8)) * DKH];

            acc = __builtin_amdgcn_wmma_f32_16x16x32_f16(false, a,
                                                         false, b,
                                                         (short)0, acc,
                                                         false, false);
        }
        float* crow = ctx + ((size_t)bh * SEQ + q0 + m0) * DKH;
        #pragma unroll
        for (int r = 0; r < 8; ++r)
            __builtin_nontemporal_store(
                acc[r], &crow[(size_t)(sel * 8 + r) * DKH + n0 + l16]);
    }
}

extern "C" void kernel_launch(void* const* d_in, const int* in_sizes, int n_in,
                              void* d_out, int out_size, void* d_ws, size_t ws_size,
                              hipStream_t stream) {
    const int B = 4, H = 8;
    const float* Q = (const float*)d_in[0];
    const float* K = (const float*)d_in[1];
    const float* V = (const float*)d_in[2];
    (void)in_sizes; (void)n_in; (void)d_ws; (void)ws_size; (void)out_size;

    float* out  = (float*)d_out;
    float* ctx  = out;                                      // [B,H,S,DK]
    float* attn = out + (size_t)B * H * SEQ * DKH;          // [B,H,S,S]

    dim3 grid(SEQ / QT, B * H);
    attn_fwd_kernel<<<grid, 256, 0, stream>>>(Q, K, V, ctx, attn);
}
